// VanillaMambaBlock_79791902425749
// MI455X (gfx1250) — compile-verified, matrix operand order checked by reading, not run on hardware
//
#include <hip/hip_runtime.h>
#include <hip/hip_bf16.h>
#include <math.h>

typedef __attribute__((ext_vector_type(16))) __bf16 v16bf;
typedef __attribute__((ext_vector_type(8)))  __bf16 v8bf;
typedef __attribute__((ext_vector_type(8)))  float  v8f;

__device__ __forceinline__ __bf16 f2bf(float f) {
  union { float f; unsigned u; } v; v.f = f;
  unsigned u = v.u;
  unsigned r = (u + 0x7FFFu + ((u >> 16) & 1u)) >> 16;   // round-to-nearest-even
  union { unsigned short s; __bf16 b; } o; o.s = (unsigned short)r;
  return o.b;
}

// -------- f32 -> bf16 conversion (once per operand, outside the GEMM loop) --------
__global__ void cvt_bf16_kernel(const float* __restrict__ src, __bf16* __restrict__ dst, int n) {
  int i = blockIdx.x * blockDim.x + threadIdx.x;
  if (i < n) dst[i] = f2bf(src[i]);
}

// Fragment: elements 0..7 = K[koff..koff+7], 8..15 = K[16+koff..16+koff+7]
// (CDNA5 ISA 16-bit A/B layout). p points at (row, kbase+koff); two b128 loads.
__device__ __forceinline__ v16bf load_frag_bf(const __bf16* __restrict__ p) {
  v8bf lo = *(const v8bf*)(p);
  v8bf hi = *(const v8bf*)(p + 16);
  return __builtin_shufflevector(lo, hi, 0,1,2,3,4,5,6,7,8,9,10,11,12,13,14,15);
}

// Load one full fragment set (A row-strip + 4 B tiles) at K-offset k.
__device__ __forceinline__ void load_set(const __bf16* __restrict__ ap,
                                         const __bf16* __restrict__ b0,
                                         const __bf16* __restrict__ b1,
                                         const __bf16* __restrict__ b2,
                                         const __bf16* __restrict__ b3,
                                         int k,
                                         v16bf& A, v16bf& B0, v16bf& B1, v16bf& B2, v16bf& B3) {
  __builtin_prefetch(ap + k + 64, 0, 1);   // speculative global_prefetch_b8
  A  = load_frag_bf(ap + k);
  B0 = load_frag_bf(b0 + k);
  B1 = load_frag_bf(b1 + k);
  B2 = load_frag_bf(b2 + k);
  B3 = load_frag_bf(b3 + k);
}

// out[M,N] = act( A[M,K](lda, bf16) @ W[N,K]^T(bf16) + bias )
// REQUIRES N % 64 == 0 and K % 64 == 0 (callers guarantee).
// Double-buffered pipeline: WMMAs on one fragment set overlap loads of the other.
// act: 0 none, 1 softplus, 2 exact gelu. out (f32) / out_bf (bf16) nullable.
__global__ void __launch_bounds__(128)
gemm_wmma_bf16(const __bf16* __restrict__ A, int lda,
               const __bf16* __restrict__ W,
               const float* __restrict__ bias,
               float* __restrict__ out,
               __bf16* __restrict__ out_bf,
               int M, int N, int K, int act)
{
  const int lane    = threadIdx.x & 31;
  const int wave    = threadIdx.x >> 5;
  const int rowBase = blockIdx.y * 64 + wave * 16;
  const int colBase = blockIdx.x * 64;
  if (rowBase >= M) return;

  const int r    = lane & 15;
  const int half = lane >> 4;
  const int koff = half * 8;

  v8f acc0 = (v8f){0.f,0.f,0.f,0.f,0.f,0.f,0.f,0.f};
  v8f acc1 = acc0, acc2 = acc0, acc3 = acc0;

  const __bf16* ap  = A + (size_t)(rowBase + r) * lda + koff;
  const __bf16* bp0 = W + (size_t)(colBase +  0 + r) * K + koff;
  const __bf16* bp1 = W + (size_t)(colBase + 16 + r) * K + koff;
  const __bf16* bp2 = W + (size_t)(colBase + 32 + r) * K + koff;
  const __bf16* bp3 = W + (size_t)(colBase + 48 + r) * K + koff;

  v16bf aX, b0X, b1X, b2X, b3X;   // buffer X
  v16bf aY, b0Y, b1Y, b2Y, b3Y;   // buffer Y

  load_set(ap, bp0, bp1, bp2, bp3, 0, aX, b0X, b1X, b2X, b3X);

  int k0 = 0;
  for (; k0 + 64 < K; k0 += 64) {
    // issue loads for k0+32 while computing k0
    load_set(ap, bp0, bp1, bp2, bp3, k0 + 32, aY, b0Y, b1Y, b2Y, b3Y);
    acc0 = __builtin_amdgcn_wmma_f32_16x16x32_bf16(false, aX, false, b0X, (short)0, acc0, false, false);
    acc1 = __builtin_amdgcn_wmma_f32_16x16x32_bf16(false, aX, false, b1X, (short)0, acc1, false, false);
    acc2 = __builtin_amdgcn_wmma_f32_16x16x32_bf16(false, aX, false, b2X, (short)0, acc2, false, false);
    acc3 = __builtin_amdgcn_wmma_f32_16x16x32_bf16(false, aX, false, b3X, (short)0, acc3, false, false);
    // issue loads for k0+64 while computing k0+32
    load_set(ap, bp0, bp1, bp2, bp3, k0 + 64, aX, b0X, b1X, b2X, b3X);
    acc0 = __builtin_amdgcn_wmma_f32_16x16x32_bf16(false, aY, false, b0Y, (short)0, acc0, false, false);
    acc1 = __builtin_amdgcn_wmma_f32_16x16x32_bf16(false, aY, false, b1Y, (short)0, acc1, false, false);
    acc2 = __builtin_amdgcn_wmma_f32_16x16x32_bf16(false, aY, false, b2Y, (short)0, acc2, false, false);
    acc3 = __builtin_amdgcn_wmma_f32_16x16x32_bf16(false, aY, false, b3Y, (short)0, acc3, false, false);
  }
  // tail: chunks k0 and k0+32
  load_set(ap, bp0, bp1, bp2, bp3, k0 + 32, aY, b0Y, b1Y, b2Y, b3Y);
  acc0 = __builtin_amdgcn_wmma_f32_16x16x32_bf16(false, aX, false, b0X, (short)0, acc0, false, false);
  acc1 = __builtin_amdgcn_wmma_f32_16x16x32_bf16(false, aX, false, b1X, (short)0, acc1, false, false);
  acc2 = __builtin_amdgcn_wmma_f32_16x16x32_bf16(false, aX, false, b2X, (short)0, acc2, false, false);
  acc3 = __builtin_amdgcn_wmma_f32_16x16x32_bf16(false, aX, false, b3X, (short)0, acc3, false, false);
  acc0 = __builtin_amdgcn_wmma_f32_16x16x32_bf16(false, aY, false, b0Y, (short)0, acc0, false, false);
  acc1 = __builtin_amdgcn_wmma_f32_16x16x32_bf16(false, aY, false, b1Y, (short)0, acc1, false, false);
  acc2 = __builtin_amdgcn_wmma_f32_16x16x32_bf16(false, aY, false, b2Y, (short)0, acc2, false, false);
  acc3 = __builtin_amdgcn_wmma_f32_16x16x32_bf16(false, aY, false, b3Y, (short)0, acc3, false, false);

  // D layout: lanes 0-15 -> M=rowBase+j; lanes 16-31 -> M=rowBase+8+j; N=colBase+t*16+r
  const int rbase = rowBase + half * 8;
  v8f accs[4] = {acc0, acc1, acc2, acc3};
  #pragma unroll
  for (int t = 0; t < 4; ++t) {
    int col  = colBase + t * 16 + r;
    float bv = bias ? bias[col] : 0.f;
    #pragma unroll
    for (int j = 0; j < 8; ++j) {
      float v = accs[t][j] + bv;
      if (act == 1)      v = (v > 20.f) ? v : log1pf(__expf(v));                 // softplus
      else if (act == 2) v = 0.5f * v * (1.f + erff(v * 0.70710678118654752f));  // gelu
      size_t o = (size_t)(rbase + j) * N + col;
      if (out)    out[o]    = v;
      if (out_bf) out_bf[o] = f2bf(v);
    }
  }
}

// u = silu(depthwise causal conv(xi) + b), xi = xz[..., :2048]; dual f32/bf16 output
__global__ void conv_silu_kernel(const float* __restrict__ xz,
                                 const float* __restrict__ cw,
                                 const float* __restrict__ cb,
                                 float* __restrict__ u,
                                 __bf16* __restrict__ u_bf)
{
  const int DI = 2048, L = 1024;
  int idx = blockIdx.x * blockDim.x + threadIdx.x;
  if (idx >= 2 * L * DI) return;
  int d  = idx % DI;
  int bl = idx / DI;
  int l  = bl % L;
  int b  = bl / L;
  float s = cb[d];
  #pragma unroll
  for (int j = 0; j < 4; ++j) {
    int ls = l - 3 + j;
    if (ls >= 0) s += cw[d * 4 + j] * xz[((size_t)(b * L + ls)) * 4096 + d];
  }
  float v = s / (1.f + __expf(-s));
  u[(size_t)bl * DI + d]    = v;
  u_bf[(size_t)bl * DI + d] = f2bf(v);
}

// Sequential selective scan; one thread per (b,d); emits y gated by silu(z) in bf16.
// dbc rows have stride 128: [64:80)=B, [80:96)=C (cols 96..127 are padding).
__global__ void scan_kernel(const float* __restrict__ delta,
                            const float* __restrict__ u,
                            const float* __restrict__ xz,
                            const float* __restrict__ dbc,
                            const float* __restrict__ A_log,
                            const float* __restrict__ Dvec,
                            __bf16* __restrict__ y_bf)
{
  const int DI = 2048, L = 1024, NS = 16;
  int t = blockIdx.x * blockDim.x + threadIdx.x;
  if (t >= 2 * DI) return;
  int d = t % DI;
  int b = t / DI;

  float a[NS], h[NS];
  #pragma unroll
  for (int n = 0; n < NS; ++n) { a[n] = -__expf(A_log[d * NS + n]); h[n] = 0.f; }
  float Dd = Dvec[d];

  for (int l = 0; l < L; ++l) {
    size_t row = (size_t)(b * L + l);
    float dt = delta[row * DI + d];
    float uu = u[row * DI + d];
    float zz = xz[row * 4096 + 2048 + d];
    const float* bc = dbc + row * 128;
    float yy = 0.f;
    #pragma unroll
    for (int n = 0; n < NS; ++n) {
      float dA = __expf(dt * a[n]);
      h[n] = dA * h[n] + dt * bc[64 + n] * uu;
      yy  += h[n] * bc[80 + n];
    }
    yy += uu * Dd;
    yy *= zz / (1.f + __expf(-zz));
    y_bf[row * DI + d] = f2bf(yy);
  }
}

static inline void cvt(const float* s, __bf16* d, int n, hipStream_t st) {
  cvt_bf16_kernel<<<(n + 255) / 256, 256, 0, st>>>(s, d, n);
}

extern "C" void kernel_launch(void* const* d_in, const int* in_sizes, int n_in,
                              void* d_out, int out_size, void* d_ws, size_t ws_size,
                              hipStream_t stream)
{
  const float* x         = (const float*)d_in[0];
  const float* in_proj_w = (const float*)d_in[1];
  const float* conv_w    = (const float*)d_in[2];
  const float* conv_b    = (const float*)d_in[3];
  const float* x_proj_w  = (const float*)d_in[4];
  const float* dt_proj_w = (const float*)d_in[5];
  const float* dt_proj_b = (const float*)d_in[6];
  const float* A_log     = (const float*)d_in[7];
  const float* Dv        = (const float*)d_in[8];
  const float* out_proj  = (const float*)d_in[9];
  const float* ff_w1     = (const float*)d_in[10];
  const float* ff_b1     = (const float*)d_in[11];
  const float* ff_w2     = (const float*)d_in[12];
  const float* ff_b2     = (const float*)d_in[13];

  const int M = 2048;                    // B*L tokens
  size_t off = 0;
  char* base = (char*)d_ws;
  auto alloc = [&](size_t bytes) -> void* {
    void* p = base + off;
    off += (bytes + 255) & ~(size_t)255;
    return p;
  };

  // f32 scratch
  float* xz    = (float*)alloc((size_t)M * 4096 * 4);
  float* u     = (float*)alloc((size_t)M * 2048 * 4);
  float* dbc   = (float*)alloc((size_t)M * 128  * 4);   // padded N=128
  float* delta = (float*)alloc((size_t)M * 2048 * 4);
  // bf16 scratch
  __bf16* x_bf    = (__bf16*)alloc((size_t)M * 1024 * 2);
  __bf16* w_in_bf = (__bf16*)alloc((size_t)4096 * 1024 * 2);
  __bf16* u_bf    = (__bf16*)alloc((size_t)M * 2048 * 2);
  __bf16* w_xp_bf = (__bf16*)alloc((size_t)128 * 2048 * 2); // rows 96..127 padding (never read back)
  __bf16* dbc_bf  = (__bf16*)alloc((size_t)M * 128 * 2);    // padded N=128
  __bf16* w_dt_bf = (__bf16*)alloc((size_t)2048 * 64 * 2);
  __bf16* y_bf    = (__bf16*)alloc((size_t)M * 2048 * 2);
  __bf16* w_op_bf = (__bf16*)alloc((size_t)1024 * 2048 * 2);
  __bf16* w_f1_bf = (__bf16*)alloc((size_t)4096 * 1024 * 2);
  __bf16* w_f2_bf = (__bf16*)alloc((size_t)1024 * 4096 * 2);
  // aliases (source regions dead after the scan)
  __bf16* mo_bf = u_bf;            // [2048,1024] <= u_bf capacity
  __bf16* h1_bf = (__bf16*)xz;     // [2048,4096] <= xz capacity

  // 0) one-time f32->bf16 conversions (weights + x)
  cvt(x,         x_bf,    M * 1024,    stream);
  cvt(in_proj_w, w_in_bf, 4096 * 1024, stream);
  cvt(x_proj_w,  w_xp_bf, 96 * 2048,   stream);   // rows 96..127 left as-is (outputs unread)
  cvt(dt_proj_w, w_dt_bf, 2048 * 64,   stream);
  cvt(out_proj,  w_op_bf, 1024 * 2048, stream);
  cvt(ff_w1,     w_f1_bf, 4096 * 1024, stream);
  cvt(ff_w2,     w_f2_bf, 1024 * 4096, stream);

  dim3 blk(128);
  // 1) in_proj: xz = x @ in_proj_w^T
  gemm_wmma_bf16<<<dim3(4096/64, 2048/64), blk, 0, stream>>>(x_bf, 1024, w_in_bf, nullptr, xz, nullptr, M, 4096, 1024, 0);
  // 2) depthwise causal conv + SiLU -> u (f32 + bf16)
  conv_silu_kernel<<<(2*1024*2048 + 255)/256, 256, 0, stream>>>(xz, conv_w, conv_b, u, u_bf);
  // 3) x_proj (padded to N=128): dbc = u @ x_proj_w^T -> f32 (scan) + bf16 (dt_proj)
  gemm_wmma_bf16<<<dim3(128/64, 2048/64), blk, 0, stream>>>(u_bf, 2048, w_xp_bf, nullptr, dbc, dbc_bf, M, 128, 2048, 0);
  // 4) dt_proj: delta = softplus(dbc[:,:64] @ dt_proj_w^T + b)   (A strided lda=128)
  gemm_wmma_bf16<<<dim3(2048/64, 2048/64), blk, 0, stream>>>(dbc_bf, 128, w_dt_bf, dt_proj_b, delta, nullptr, M, 2048, 64, 1);
  // 5) selective scan (+ D skip + SiLU(z) gate) -> y_bf
  scan_kernel<<<(4096 + 255)/256, 256, 0, stream>>>(delta, u, xz, dbc, A_log, Dv, y_bf);
  // 6) out_proj: mo_bf = y @ out_proj^T
  gemm_wmma_bf16<<<dim3(1024/64, 2048/64), blk, 0, stream>>>(y_bf, 2048, w_op_bf, nullptr, nullptr, mo_bf, M, 1024, 2048, 0);
  // 7) ff1: h1_bf = gelu(mo @ ff_w1^T + b1)
  gemm_wmma_bf16<<<dim3(4096/64, 2048/64), blk, 0, stream>>>(mo_bf, 1024, w_f1_bf, ff_b1, nullptr, h1_bf, M, 4096, 1024, 2);
  // 8) ff2: out = h1 @ ff_w2^T + b2
  gemm_wmma_bf16<<<dim3(1024/64, 2048/64), blk, 0, stream>>>(h1_bf, 4096, w_f2_bf, ff_b2, (float*)d_out, nullptr, M, 1024, 4096, 0);
}
